// MeshDeformation_61821759258844
// MI455X (gfx1250) — matrix operand order repaired and checked by reading, not compiled, not run on hardware
//
#include <hip/hip_runtime.h>
#include <hip/hip_bf16.h>

// ---- problem constants (match reference) ----
#define NVERT 6890
#define BATCH 8
#define NEDGE 41340
#define MTOT  55120      // BATCH*NVERT == 3445*16 (odd number of 16-row tiles)
#define HIDD  256
#define OUTD  3

typedef __attribute__((ext_vector_type(16))) __bf16        v16bf;
typedef __attribute__((ext_vector_type(16))) unsigned short v16u;
typedef __attribute__((ext_vector_type(8)))  unsigned short v8u;
typedef __attribute__((ext_vector_type(8)))  float          v8f;

__device__ __forceinline__ unsigned short f2bf(float f) {
    // round-to-nearest-even fp32 -> bf16
    unsigned int u = __builtin_bit_cast(unsigned int, f);
    u += 0x7fffu + ((u >> 16) & 1u);
    return (unsigned short)(u >> 16);
}

// ---------------------------------------------------------------------------
// Weight swizzle: W [K][256] f32  ->  bf16 fragments in WMMA 16x16x32 B-layout
// fragment index = ((ntile*KSteps + kk)*32 + lane), 16 bf16 per lane:
//   col   = ntile*16 + (lane&15)
//   kbase = kk*32 + (lane>>4)*8 ;  element j -> W[kbase+j+(j&8)][col]
// ---------------------------------------------------------------------------
__global__ __launch_bounds__(128)
void wconv_kernel(const float* __restrict__ W, unsigned short* __restrict__ Wf, int K)
{
    const int kst = K / 32;
    const int total = 16 * kst * 32;
    int idx = blockIdx.x * 128 + threadIdx.x;
    if (idx >= total) return;
    const int lane  = idx & 31;
    const int kk    = (idx >> 5) % kst;
    const int ntile = idx / (32 * kst);
    const int col   = ntile * 16 + (lane & 15);
    const int kbase = kk * 32 + (lane >> 4) * 8;
#pragma unroll
    for (int j = 0; j < 16; ++j)
        Wf[(size_t)idx * 16 + j] = f2bf(W[(size_t)(kbase + j + (j & 8)) * HIDD + col]);
}

// ---------------------------------------------------------------------------
// Dual GEMM:  T = X@W ,  ACC = X@L + bias      (X: [MTOT,K] f32, out N=256)
// Block = 128 threads (4 waves), one 32-row M tile per block (2 subtiles).
// Each wave owns a 64-column slice; every B fragment loaded once feeds TWO
// WMMAs (both M subtiles), halving the dominant L2 weight-fragment traffic.
// ---------------------------------------------------------------------------
template <int K>
__global__ __launch_bounds__(128)
void gemm_dual_kernel(const float* __restrict__ X,
                      const unsigned short* __restrict__ WfW,
                      const unsigned short* __restrict__ WfL,
                      const float* __restrict__ bias,
                      float* __restrict__ T, float* __restrict__ ACC)
{
    constexpr int KST  = K / 32;
    constexpr int LSTR = K + 8;                 // padded LDS row stride (shorts)
    __shared__ unsigned short As[32 * LSTR];

    const int tid   = threadIdx.x;
    const int lane  = tid & 31;
    const int wave  = tid >> 5;                 // 0..3
    const size_t rowBase = (size_t)blockIdx.x * 32;

    // stage 32 x K fp32 rows -> bf16 in LDS (zero-fill past MTOT)
    const float4* Xv = (const float4*)X;
    for (int f = tid; f < 32 * K / 4; f += 128) {
        const int r  = f / (K / 4);
        const int c4 = f % (K / 4);
        const size_t gr = rowBase + r;
        float4 v = make_float4(0.f, 0.f, 0.f, 0.f);
        if (gr < MTOT) v = Xv[gr * (K / 4) + c4];
        unsigned short* p = &As[r * LSTR + c4 * 4];
        p[0] = f2bf(v.x); p[1] = f2bf(v.y); p[2] = f2bf(v.z); p[3] = f2bf(v.w);
    }
    __syncthreads();

    v8f accW0[4], accW1[4], accL0[4], accL1[4];
#pragma unroll
    for (int t = 0; t < 4; ++t)
#pragma unroll
        for (int j = 0; j < 8; ++j) {
            accW0[t][j] = 0.f; accW1[t][j] = 0.f;
            accL0[t][j] = 0.f; accL1[t][j] = 0.f;
        }

    const int row   = lane & 15;
    const int khalf = lane >> 4;

    for (int kk = 0; kk < KST; ++kk) {
        // A fragments for both M subtiles (16-bit A layout: lanes 0-15 hold
        // K 0..7/16..23, lanes 16-31 hold K 8..15/24..31 of their row)
        const int kb = kk * 32 + khalf * 8;
        v8u a0lo = *(const v8u*)&As[row * LSTR + kb];
        v8u a0hi = *(const v8u*)&As[row * LSTR + kb + 16];
        v8u a1lo = *(const v8u*)&As[(row + 16) * LSTR + kb];
        v8u a1hi = *(const v8u*)&As[(row + 16) * LSTR + kb + 16];
        v16u a0u, a1u;
#pragma unroll
        for (int j = 0; j < 8; ++j) {
            a0u[j] = a0lo[j]; a0u[8 + j] = a0hi[j];
            a1u[j] = a1lo[j]; a1u[8 + j] = a1hi[j];
        }
        v16bf a0 = __builtin_bit_cast(v16bf, a0u);
        v16bf a1 = __builtin_bit_cast(v16bf, a1u);

        if (kk + 1 < KST) {
            size_t pf = ((size_t)((wave * 4) * KST + kk + 1) * 32 + lane) * 16;
            __builtin_prefetch(WfW + pf, 0, 1);   // global_prefetch_b8
            __builtin_prefetch(WfL + pf, 0, 1);
        }

#pragma unroll
        for (int t = 0; t < 4; ++t) {
            const int ntile = wave * 4 + t;
            const size_t off = ((size_t)(ntile * KST + kk) * 32 + lane) * 16;
            v16bf bw = __builtin_bit_cast(v16bf, *(const v16u*)(WfW + off));
            accW0[t] = __builtin_amdgcn_wmma_f32_16x16x32_bf16(
                false, a0, false, bw, (short)0, accW0[t], false, false);
            accW1[t] = __builtin_amdgcn_wmma_f32_16x16x32_bf16(
                false, a1, false, bw, (short)0, accW1[t], false, false);
            v16bf bl = __builtin_bit_cast(v16bf, *(const v16u*)(WfL + off));
            accL0[t] = __builtin_amdgcn_wmma_f32_16x16x32_bf16(
                false, a0, false, bl, (short)0, accL0[t], false, false);
            accL1[t] = __builtin_amdgcn_wmma_f32_16x16x32_bf16(
                false, a1, false, bl, (short)0, accL1[t], false, false);
        }
    }

    // C/D layout: VGPR v -> M = v + (lane>=16)*8, N = lane&15
#pragma unroll
    for (int t = 0; t < 4; ++t) {
        const int col = wave * 64 + t * 16 + (lane & 15);
        const float bv = bias[col];
#pragma unroll
        for (int v = 0; v < 8; ++v) {
            const size_t r0 = rowBase + v + khalf * 8;   // always < MTOT
            const size_t r1 = r0 + 16;                   // may overflow on last block
            T[r0 * HIDD + col]   = accW0[t][v];
            ACC[r0 * HIDD + col] = accL0[t][v] + bv;
            if (r1 < MTOT) {
                T[r1 * HIDD + col]   = accW1[t][v];
                ACC[r1 * HIDD + col] = accL1[t][v] + bv;
            }
        }
    }
}

// ---------------------------------------------------------------------------
// SpMM scatter:  ACC[b,dst,:] += val * T[b,src,:]   (f32 global atomics)
// one thread = (edge, 4-feature chunk), loops over batch
// ---------------------------------------------------------------------------
__global__ __launch_bounds__(256)
void spmm_kernel(const float* __restrict__ T, float* __restrict__ ACC,
                 const int* __restrict__ src, const int* __restrict__ dst,
                 const float* __restrict__ val)
{
    int idx = blockIdx.x * 256 + threadIdx.x;
    if (idx >= NEDGE * (HIDD / 4)) return;
    const int e = idx >> 6;
    const int c = (idx & 63) * 4;
    const int s = src[e], d = dst[e];
    const float v = val[e];
#pragma unroll
    for (int b = 0; b < BATCH; ++b) {
        const size_t so = ((size_t)b * NVERT + s) * HIDD + c;
        const size_t dofs = ((size_t)b * NVERT + d) * HIDD + c;
        float4 tv = *(const float4*)(T + so);
        atomicAdd(ACC + dofs + 0, v * tv.x);
        atomicAdd(ACC + dofs + 1, v * tv.y);
        atomicAdd(ACC + dofs + 2, v * tv.z);
        atomicAdd(ACC + dofs + 3, v * tv.w);
    }
}

// ---- elementwise epilogues (float4) ----
__global__ __launch_bounds__(256)
void relu_kernel(const float* __restrict__ A, float* __restrict__ X, int n4)
{
    int i = blockIdx.x * 256 + threadIdx.x;
    if (i >= n4) return;
    float4 a = ((const float4*)A)[i];
    a.x = fmaxf(a.x, 0.f); a.y = fmaxf(a.y, 0.f);
    a.z = fmaxf(a.z, 0.f); a.w = fmaxf(a.w, 0.f);
    ((float4*)X)[i] = a;
}

__global__ __launch_bounds__(256)
void resid_kernel(const float* __restrict__ A, float* __restrict__ X, int n4)
{
    int i = blockIdx.x * 256 + threadIdx.x;
    if (i >= n4) return;
    float4 a = ((const float4*)A)[i];
    float4 x = ((float4*)X)[i];
    x.x = (x.x + fmaxf(a.x, 0.f)) * 0.5f;
    x.y = (x.y + fmaxf(a.y, 0.f)) * 0.5f;
    x.z = (x.z + fmaxf(a.z, 0.f)) * 0.5f;
    x.w = (x.w + fmaxf(a.w, 0.f)) * 0.5f;
    ((float4*)X)[i] = x;
}

// ---- final 256 -> 3 conv: dense part (VALU; N=3 is too thin for WMMA) ----
__global__ __launch_bounds__(256)
void dense3_kernel(const float* __restrict__ X, const float* __restrict__ W2,
                   const float* __restrict__ L2, const float* __restrict__ b2,
                   float* __restrict__ T2, float* __restrict__ A2)
{
    int m = blockIdx.x * 256 + threadIdx.x;
    if (m >= MTOT) return;
    const float* x = X + (size_t)m * HIDD;
    float aw[3] = {0.f, 0.f, 0.f}, al[3] = {0.f, 0.f, 0.f};
    for (int k = 0; k < HIDD; ++k) {
        float xv = x[k];
#pragma unroll
        for (int j = 0; j < 3; ++j) {
            aw[j] = fmaf(xv, W2[k * 3 + j], aw[j]);
            al[j] = fmaf(xv, L2[k * 3 + j], al[j]);
        }
    }
#pragma unroll
    for (int j = 0; j < 3; ++j) {
        T2[(size_t)m * 3 + j] = aw[j];
        A2[(size_t)m * 3 + j] = al[j] + b2[j];
    }
}

__global__ __launch_bounds__(256)
void spmm3_kernel(const float* __restrict__ T2, float* __restrict__ A2,
                  const int* __restrict__ src, const int* __restrict__ dst,
                  const float* __restrict__ val)
{
    int e = blockIdx.x * 256 + threadIdx.x;
    if (e >= NEDGE) return;
    const int s = src[e], d = dst[e];
    const float v = val[e];
#pragma unroll
    for (int b = 0; b < BATCH; ++b) {
        const size_t so = ((size_t)b * NVERT + s) * 3;
        const size_t dofs = ((size_t)b * NVERT + d) * 3;
#pragma unroll
        for (int j = 0; j < 3; ++j)
            atomicAdd(A2 + dofs + j, v * T2[so + j]);
    }
}

__global__ __launch_bounds__(256)
void final_kernel(const float* __restrict__ A2, float* __restrict__ out, int n)
{
    int i = blockIdx.x * 256 + threadIdx.x;
    if (i >= n) return;
    out[i] = tanhf(A2[i]) * 0.1f;
}

// ---------------------------------------------------------------------------
extern "C" void kernel_launch(void* const* d_in, const int* in_sizes, int n_in,
                              void* d_out, int out_size, void* d_ws, size_t ws_size,
                              hipStream_t stream)
{
    (void)in_sizes; (void)n_in; (void)out_size; (void)ws_size;
    const float* verts = (const float*)d_in[0];
    const int*   esrc  = (const int*)d_in[1];
    const int*   edst  = (const int*)d_in[2];
    const float* eval_ = (const float*)d_in[3];
    const float* W1    = (const float*)d_in[4];
    const float* L1    = (const float*)d_in[5];
    const float* b1    = (const float*)d_in[6];
    const float* Wb    = (const float*)d_in[7];
    const float* Lb    = (const float*)d_in[8];
    const float* bb    = (const float*)d_in[9];
    const float* W2    = (const float*)d_in[10];
    const float* L2    = (const float*)d_in[11];
    const float* b2    = (const float*)d_in[12];

    char* ws = (char*)d_ws;
    const size_t SZ = (size_t)MTOT * HIDD * sizeof(float);    // 56.44 MB
    float* T    = (float*)(ws);
    float* ACC  = (float*)(ws + SZ);
    float* Xc   = (float*)(ws + 2 * SZ);
    float* H    = (float*)(ws + 3 * SZ);
    unsigned short* WfW = (unsigned short*)(ws + 4 * SZ);
    unsigned short* WfL = (unsigned short*)(ws + 4 * SZ + (size_t)HIDD * HIDD * 2);
    float* T2   = (float*)(ws + 4 * SZ + 2 * (size_t)HIDD * HIDD * 2);
    float* A2   = T2 + (size_t)MTOT * 3;

    const int n4 = MTOT * HIDD / 4;
    const int spmmThreads = NEDGE * (HIDD / 4);
    const dim3 gemmGrid((MTOT + 31) / 32);     // 1723 blocks of 32 rows

    auto conv = [&](const float* Xin, const float* Wp, const float* Lp,
                    const float* bp, int K) {
        const int wtot = 16 * (K / 32) * 32;
        wconv_kernel<<<(wtot + 127) / 128, 128, 0, stream>>>(Wp, WfW, K);
        wconv_kernel<<<(wtot + 127) / 128, 128, 0, stream>>>(Lp, WfL, K);
        if (K == 128)
            gemm_dual_kernel<128><<<gemmGrid, 128, 0, stream>>>(Xin, WfW, WfL, bp, T, ACC);
        else
            gemm_dual_kernel<256><<<gemmGrid, 128, 0, stream>>>(Xin, WfW, WfL, bp, T, ACC);
        spmm_kernel<<<(spmmThreads + 255) / 256, 256, 0, stream>>>(T, ACC, esrc, edst, eval_);
    };

    // conv1: 128 -> 256, then ReLU
    conv(verts, W1, L1, b1, 128);
    relu_kernel<<<(n4 + 255) / 256, 256, 0, stream>>>(ACC, Xc, n4);

    // 4 residual blocks
    for (int i = 0; i < 4; ++i) {
        const size_t w0 = (size_t)(2 * i) * HIDD * HIDD;
        const size_t w1 = (size_t)(2 * i + 1) * HIDD * HIDD;
        conv(Xc, Wb + w0, Lb + w0, bb + (size_t)(2 * i) * HIDD, 256);
        relu_kernel<<<(n4 + 255) / 256, 256, 0, stream>>>(ACC, H, n4);
        conv(H, Wb + w1, Lb + w1, bb + (size_t)(2 * i + 1) * HIDD, 256);
        resid_kernel<<<(n4 + 255) / 256, 256, 0, stream>>>(ACC, Xc, n4);
    }

    // conv2: 256 -> 3, tanh * 0.1
    dense3_kernel<<<(MTOT + 255) / 256, 256, 0, stream>>>(Xc, W2, L2, b2, T2, A2);
    spmm3_kernel<<<(NEDGE + 255) / 256, 256, 0, stream>>>(T2, A2, esrc, edst, eval_);
    final_kernel<<<(MTOT * 3 + 255) / 256, 256, 0, stream>>>(A2, (float*)d_out, MTOT * 3);
}
